// StructureModule_51591147160221
// MI455X (gfx1250) — matrix operand order repaired and checked by reading, not compiled, not run on hardware
//
#include <hip/hip_runtime.h>

// ============================================================================
// AlphaFold-style Structure Module forward pass for MI455X (gfx1250, wave32).
// All dense _conv1 GEMMs run through a f16-WMMA kernel
// (v_wmma_f32_16x16x32_f16, fp32 accumulate), templated on GUARD:
//  - GUARD=false (hot path, dims tile-aligned): float4 staging loads, no
//    predication anywhere.
//  - GUARD=true (edge dims): clamped addresses + multiplicative {0,1} mask so
//    loads stay unconditional (no exec-mask diamonds).
// Everything else is VALU. Memory-bound on z (75.5MB): ~0.6GB traffic
// => ~25us at 23.3 TB/s HBM.
// ============================================================================

typedef __attribute__((ext_vector_type(16))) _Float16 v16h;
typedef __attribute__((ext_vector_type(8)))  float    v8f;

#define NRES   384
#define NN     (NRES * NRES)
#define CS     384
#define CZ     128
#define NHEAD  12
#define CIPA   16
#define NQP    4
#define NPV    8
#define CA     128
#define INS    2017            // CZ*H + CIPA*H + 3*H*NPV + 1
#define NLAYER 4

#define LN_EPS        1e-5f
#define W_C_HALF      0.11785113019775793f   // 0.5*sqrt(2/(9*NQP))
#define W_L           0.57735026918962576f   // sqrt(1/3)
#define INV_SQRT_CIPA 0.25f

// cat row offsets (channel-major [ch][n])
#define CAT_O1  0
#define CAT_O2  (1536 * NRES)
#define CAT_O3  (1728 * NRES)
#define CAT_O3N (2016 * NRES)

// loss accumulator slots in ws
#define ACC_F0   0   // fape numerator, per layer (4)
#define ACC_T0   4   // torsion numerator (lt + 0.02*la), per layer (4)
#define ACC_MSUM 8

// ---------------------------------------------------------------------------
// WMMA GEMM: C[M,N] = act( A[M,K] * B[K,N] + bias[M] + resid[M,N] )
// Block = 128 threads = 4 waves; macro tile 64(M) x 16(N), K-step 32.
// ---------------------------------------------------------------------------
#define GEMM_BM 64
#define GEMM_BN 16
#define GEMM_BK 32
#define LDA_PAD 4     // A row stride 36 floats: 16B-aligned, bank-conflict-free
#define LDB_PAD 4     // B row stride 20 floats

template <bool GUARD>
__global__ __launch_bounds__(128)
void gemm_wmma_kernel(const float* __restrict__ A,
                      const float* __restrict__ B,
                      const float* __restrict__ bias,
                      const float* __restrict__ resid,
                      float* __restrict__ C,
                      int M, int N, int K,
                      int reluB, int reluOut)
{
    __shared__ float lds_a[GEMM_BM][GEMM_BK + LDA_PAD];
    __shared__ float lds_b[GEMM_BK][GEMM_BN + LDB_PAD];

    const int tid  = threadIdx.x;        // 0..127
    const int wave = tid >> 5;           // 0..3 -> M sub-tile
    const int lane = tid & 31;
    const int hgrp = lane >> 4;          // lane half
    const int r    = lane & 15;

    const int tm_blk = blockIdx.y * GEMM_BM;
    const int tn     = blockIdx.x * GEMM_BN;
    const int tm     = tm_blk + wave * 16;

    v8f acc = {0.f, 0.f, 0.f, 0.f, 0.f, 0.f, 0.f, 0.f};

    for (int k0 = 0; k0 < K; k0 += GEMM_BK) {
        if constexpr (!GUARD) {
            // ---- A tile: 64x32 floats = 512 float4, 4 per thread (b128 loads)
            #pragma unroll
            for (int e = 0; e < 4; ++e) {
                int idx = tid + e * 128;         // 0..511
                int mm  = idx >> 3;
                int q4  = (idx & 7) * 4;
                float4 va = *(const float4*)(A + (long)(tm_blk + mm) * K + k0 + q4);
                *(float4*)&lds_a[mm][q4] = va;
            }
            // ---- B tile: 32x16 floats = 128 float4, 1 per thread
            {
                int kk = tid >> 2;
                int n4 = (tid & 3) * 4;
                float4 vb = *(const float4*)(B + (long)(k0 + kk) * N + tn + n4);
                if (reluB) {
                    vb.x = fmaxf(vb.x, 0.f); vb.y = fmaxf(vb.y, 0.f);
                    vb.z = fmaxf(vb.z, 0.f); vb.w = fmaxf(vb.w, 0.f);
                }
                *(float4*)&lds_b[kk][n4] = vb;
            }
        } else {
            // ---- guarded: clamp address, multiply by {0,1} mask.
            // load * mask cannot be folded away (0*NaN != 0), so the global
            // loads remain unconditional: no exec-mask diamonds.
            #pragma unroll
            for (int e = 0; e < (GEMM_BM * GEMM_BK) / 128; ++e) {
                int idx = tid + e * 128;
                int mm  = idx >> 5;
                int kk  = idx & 31;
                int gm  = tm_blk + mm;
                int gk  = k0 + kk;
                int cm  = gm < M ? gm : M - 1;
                int ck  = gk < K ? gk : K - 1;
                float msk = (gm < M && gk < K) ? 1.f : 0.f;
                lds_a[mm][kk] = A[(long)cm * K + ck] * msk;
            }
            #pragma unroll
            for (int e = 0; e < (GEMM_BK * GEMM_BN) / 128; ++e) {
                int idx = tid + e * 128;
                int kk  = idx >> 4;
                int nn  = idx & 15;
                int gk  = k0 + kk;
                int gn  = tn + nn;
                int ck  = gk < K ? gk : K - 1;
                int cn  = gn < N ? gn : N - 1;
                float msk = (gk < K && gn < N) ? 1.f : 0.f;
                float val = B[(long)ck * N + cn] * msk;
                if (reluB) val = fmaxf(val, 0.f);
                lds_b[kk][nn] = val;
            }
        }
        __syncthreads();

        // ---- fragments from LDS (ISA VGPR layouts), then WMMA
        v16h av, bv;
        #pragma unroll
        for (int v = 0; v < 8; ++v) {
            // A (16x32 f16): lanes 0-15: VGPR0-3 K=0..7, VGPR4-7 K=16..23;
            //                lanes 16-31: +8
            int ka = ((v & 4) ? 16 : 0) + hgrp * 8 + (v & 3) * 2;
            float2 ap = *(const float2*)&lds_a[wave * 16 + r][ka];
            av[2 * v    ] = (_Float16)ap.x;
            av[2 * v + 1] = (_Float16)ap.y;
            // B (32x16 f16): same K order as A: lanes 0-15 K=0..7,16..23; lanes 16-31: +8
            int kb = ((v & 4) ? 16 : 0) + hgrp * 8 + (v & 3) * 2;
            bv[2 * v    ] = (_Float16)lds_b[kb][r];
            bv[2 * v + 1] = (_Float16)lds_b[kb + 1][r];
        }
        acc = __builtin_amdgcn_wmma_f32_16x16x32_f16(false, av, false, bv,
                                                     (short)0, acc, false, false);
        __syncthreads();
    }

    // ---- epilogue: C/D layout: VGPR v, lane-half h -> M=v+8h, N=lane&15
    #pragma unroll
    for (int v = 0; v < 8; ++v) {
        int mo = tm + v + 8 * hgrp;
        int no = tn + r;
        if (GUARD && (mo >= M || no >= N)) continue;
        float val = acc[v];
        if (bias)  val += bias[mo];
        if (resid) val += resid[(long)mo * N + no];
        if (reluOut) val = fmaxf(val, 0.f);
        C[(long)mo * N + no] = val;
    }
}

// ---------------------------------------------------------------------------
// LayerNorm over channel axis: x[C,L] normalized over C per position l.
// Safe in-place (y == x): each thread owns one column l.
// ---------------------------------------------------------------------------
__global__ void ln_axis_kernel(const float* __restrict__ x,
                               const float* __restrict__ g,
                               const float* __restrict__ b,
                               float* __restrict__ y, int C, int L)
{
    int l = blockIdx.x * blockDim.x + threadIdx.x;
    if (l >= L) return;
    float sum = 0.f, sq = 0.f;
    for (int c = 0; c < C; ++c) {
        float v = x[(long)c * L + l];
        sum += v; sq += v * v;
    }
    float mu  = sum / (float)C;
    float var = sq / (float)C - mu * mu;
    float inv = rsqrtf(var + LN_EPS);
    for (int c = 0; c < C; ++c) {
        float v = (x[(long)c * L + l] - mu) * inv;
        y[(long)c * L + l] = v * g[c] + b[c];
    }
}

// ---------------------------------------------------------------------------
// Apply rigid transform to points: out[d,h,p,i] = R_i * pts[.,h,p,i] + t_i
// ---------------------------------------------------------------------------
__global__ void apply_frames_kernel(const float* __restrict__ pts,
                                    const float* __restrict__ t_r,
                                    const float* __restrict__ t_t,
                                    float* __restrict__ out, int P)
{
    int total = 3 * NHEAD * P * NRES;
    int idx = blockIdx.x * blockDim.x + threadIdx.x;
    if (idx >= total) return;
    int i  = idx % NRES;
    int p  = (idx / NRES) % P;
    int hh = (idx / (NRES * P)) % NHEAD;
    int d  = idx / (NRES * P * NHEAD);
    float s = t_t[i * 3 + d];
    for (int j = 0; j < 3; ++j)
        s += t_r[i * 9 + d * 3 + j] * pts[(((long)j * NHEAD + hh) * P + p) * NRES + i];
    out[(((long)d * NHEAD + hh) * P + p) * NRES + i] = s;
}

// ---------------------------------------------------------------------------
// logits[h,i,j] = W_L * (q.k/sqrt(16) + bias - softplus(gamma_h)*W_C/2*dist2)
// ---------------------------------------------------------------------------
__global__ void logits_kernel(const float* __restrict__ q,
                              const float* __restrict__ k,
                              const float* __restrict__ qg,
                              const float* __restrict__ kg,
                              const float* __restrict__ bias,
                              const float* __restrict__ gamma,
                              float* __restrict__ out)
{
    long idx = (long)blockIdx.x * blockDim.x + threadIdx.x;
    if (idx >= (long)NHEAD * NN) return;
    int j  = (int)(idx % NRES);
    int i  = (int)((idx / NRES) % NRES);
    int hh = (int)(idx / NN);
    float qk = 0.f;
    for (int c = 0; c < CIPA; ++c)
        qk += q[(c * NHEAD + hh) * NRES + i] * k[(c * NHEAD + hh) * NRES + j];
    float d2 = 0.f;
    for (int d = 0; d < 3; ++d)
        for (int p = 0; p < NQP; ++p) {
            float dd = qg[((d * NHEAD + hh) * NQP + p) * NRES + i]
                     - kg[((d * NHEAD + hh) * NQP + p) * NRES + j];
            d2 += dd * dd;
        }
    float sp = log1pf(expf(gamma[hh]));   // softplus
    out[idx] = W_L * (qk * INV_SQRT_CIPA + bias[idx] - sp * W_C_HALF * d2);
}

// ---------------------------------------------------------------------------
// row softmax over j (length 384), rows = NHEAD*NRES, one block per row
// ---------------------------------------------------------------------------
__global__ void softmax384_kernel(float* __restrict__ a)
{
    float* p = a + (long)blockIdx.x * NRES;
    int t = threadIdx.x;
    __shared__ float sm[128];
    float m = -1e30f;
    for (int j = t; j < NRES; j += 128) m = fmaxf(m, p[j]);
    sm[t] = m; __syncthreads();
    for (int s = 64; s > 0; s >>= 1) { if (t < s) sm[t] = fmaxf(sm[t], sm[t + s]); __syncthreads(); }
    m = sm[0]; __syncthreads();
    float sum = 0.f;
    for (int j = t; j < NRES; j += 128) { float e = expf(p[j] - m); p[j] = e; sum += e; }
    sm[t] = sum; __syncthreads();
    for (int s = 64; s > 0; s >>= 1) { if (t < s) sm[t] += sm[t + s]; __syncthreads(); }
    float inv = 1.f / sm[0];
    for (int j = t; j < NRES; j += 128) p[j] *= inv;
}

// o1[c,h,i] = sum_j a[h,i,j] * z[c,i,j]  -> cat channel c*H+h
__global__ void o1_kernel(const float* __restrict__ a,
                          const float* __restrict__ z,
                          float* __restrict__ cat)
{
    int idx = blockIdx.x * blockDim.x + threadIdx.x;
    if (idx >= CZ * NHEAD * NRES) return;
    int i  = idx % NRES;
    int hh = (idx / NRES) % NHEAD;
    int c  = idx / (NRES * NHEAD);
    const float* ar = a + ((long)hh * NRES + i) * NRES;
    const float* zr = z + (long)c * NN + (long)i * NRES;
    float s = 0.f;
    for (int j = 0; j < NRES; ++j) s += ar[j] * zr[j];
    cat[CAT_O1 + (c * NHEAD + hh) * NRES + i] = s;
}

// o2[c,h,i] = sum_j a[h,i,j] * v[c,h,j]  -> cat channel 1536 + c*H+h
__global__ void o2_kernel(const float* __restrict__ a,
                          const float* __restrict__ v,
                          float* __restrict__ cat)
{
    int idx = blockIdx.x * blockDim.x + threadIdx.x;
    if (idx >= CIPA * NHEAD * NRES) return;
    int i  = idx % NRES;
    int hh = (idx / NRES) % NHEAD;
    int c  = idx / (NRES * NHEAD);
    const float* ar = a + ((long)hh * NRES + i) * NRES;
    const float* vr = v + (long)(c * NHEAD + hh) * NRES;
    float s = 0.f;
    for (int j = 0; j < NRES; ++j) s += ar[j] * vr[j];
    cat[CAT_O2 + (c * NHEAD + hh) * NRES + i] = s;
}

// o3_[d,h,p,i] = sum_j a[h,i,j] * vg[d,h,p,j]
__global__ void o3a_kernel(const float* __restrict__ a,
                           const float* __restrict__ vg,
                           float* __restrict__ o3t)
{
    int idx = blockIdx.x * blockDim.x + threadIdx.x;
    if (idx >= 3 * NHEAD * NPV * NRES) return;
    int i  = idx % NRES;
    int p  = (idx / NRES) % NPV;
    int hh = (idx / (NRES * NPV)) % NHEAD;
    int d  = idx / (NRES * NPV * NHEAD);
    const float* ar = a + ((long)hh * NRES + i) * NRES;
    const float* vr = vg + (long)((d * NHEAD + hh) * NPV + p) * NRES;
    float s = 0.f;
    for (int j = 0; j < NRES; ++j) s += ar[j] * vr[j];
    o3t[((d * NHEAD + hh) * NPV + p) * NRES + i] = s;
}

// o3 = R_i^T * (o3_ - t_i)   -> cat channels 1728..2015
__global__ void o3b_kernel(const float* __restrict__ o3t,
                           const float* __restrict__ t_r,
                           const float* __restrict__ t_t,
                           float* __restrict__ cat)
{
    int idx = blockIdx.x * blockDim.x + threadIdx.x;
    if (idx >= NHEAD * NPV * NRES) return;
    int i  = idx % NRES;
    int p  = (idx / NRES) % NPV;
    int hh = idx / (NRES * NPV);
    float v3[3];
    for (int d = 0; d < 3; ++d)
        v3[d] = o3t[((d * NHEAD + hh) * NPV + p) * NRES + i] - t_t[i * 3 + d];
    for (int dp = 0; dp < 3; ++dp) {
        float s = 0.f;
        for (int d = 0; d < 3; ++d) s += t_r[i * 9 + d * 3 + dp] * v3[d];
        cat[CAT_O3 + ((dp * NHEAD + hh) * NPV + p) * NRES + i] = s;
    }
}

// o3 L2 norm over its 288 channels -> cat channel 2016
__global__ void o3n_kernel(float* __restrict__ cat)
{
    int i = blockIdx.x * blockDim.x + threadIdx.x;
    if (i >= NRES) return;
    float s = 0.f;
    for (int ch = 0; ch < 3 * NHEAD * NPV; ++ch) {
        float v = cat[CAT_O3 + ch * NRES + i];
        s += v * v;
    }
    cat[CAT_O3N + i] = sqrtf(s);
}

// t_r = I, t_t = 0
__global__ void init_frames_kernel(float* __restrict__ t_r, float* __restrict__ t_t)
{
    int n = blockIdx.x * blockDim.x + threadIdx.x;
    if (n >= NRES) return;
    for (int k = 0; k < 9; ++k) t_r[n * 9 + k] = (k == 0 || k == 4 || k == 8) ? 1.f : 0.f;
    for (int k = 0; k < 3; ++k) t_t[n * 3 + k] = 0.f;
}

// quaternion backbone update + frame composition (per residue, in place)
__global__ void bb_compose_kernel(const float* __restrict__ bu,
                                  float* __restrict__ t_r,
                                  float* __restrict__ t_t)
{
    int n = blockIdx.x * blockDim.x + threadIdx.x;
    if (n >= NRES) return;
    float bq = bu[0 * NRES + n], cq = bu[1 * NRES + n], dq = bu[2 * NRES + n];
    float ut[3] = { bu[3 * NRES + n], bu[4 * NRES + n], bu[5 * NRES + n] };
    float den = sqrtf(1.f + bq * bq + cq * cq + dq * dq);
    float aq = 1.f / den; bq /= den; cq /= den; dq /= den;
    float u[3][3] = {
        { aq*aq + bq*bq - cq*cq - dq*dq, 2*bq*cq - 2*aq*dq,             2*bq*dq + 2*aq*cq },
        { 2*bq*cq + 2*aq*dq,             aq*aq - bq*bq + cq*cq - dq*dq, 2*cq*dq - 2*aq*bq },
        { 2*bq*dq - 2*aq*cq,             2*cq*dq + 2*aq*bq,             aq*aq - bq*bq - cq*cq + dq*dq } };
    float R[3][3], T[3];
    for (int i = 0; i < 3; ++i) {
        for (int k2 = 0; k2 < 3; ++k2) {
            float s = 0.f;
            for (int j = 0; j < 3; ++j) s += t_r[n * 9 + i * 3 + j] * u[j][k2];
            R[i][k2] = s;
        }
        float s = 0.f;
        for (int j = 0; j < 3; ++j) s += t_r[n * 9 + i * 3 + j] * ut[j];
        T[i] = s + t_t[n * 3 + i];
    }
    for (int i = 0; i < 3; ++i) {
        for (int k2 = 0; k2 < 3; ++k2) t_r[n * 9 + i * 3 + k2] = R[i][k2];
        t_t[n * 3 + i] = T[i];
    }
}

// ---------------------------------------------------------------------------
// reductions / losses
// ---------------------------------------------------------------------------
__device__ float block_reduce_sum_256(float v)
{
    __shared__ float sm[256];
    int t = threadIdx.x;
    sm[t] = v; __syncthreads();
    for (int s = 128; s > 0; s >>= 1) { if (t < s) sm[t] += sm[t + s]; __syncthreads(); }
    return sm[0];
}

__global__ void zero_acc_kernel(float* __restrict__ acc)
{
    if (threadIdx.x < 32) acc[threadIdx.x] = 0.f;
}

__global__ void msum_kernel(const float* __restrict__ mask, float* __restrict__ acc)
{
    float s = 0.f;
    for (int i = threadIdx.x; i < NRES; i += 256) s += mask[i];
    s = block_reduce_sum_256(s);
    if (threadIdx.x == 0) acc[ACC_MSUM] = s;
}

// FAPE numerator: sum_ij min(10, |R_i^T(t_j - t_i) - Rt_i^T(xt_j - tt_i)|) * m_i m_j
__global__ void fape_kernel(const float* __restrict__ t_r,
                            const float* __restrict__ t_t,
                            const float* __restrict__ ttr,
                            const float* __restrict__ ttt,
                            const float* __restrict__ xt,
                            const float* __restrict__ mask,
                            float* __restrict__ acc, int layer)
{
    int idx = blockIdx.x * blockDim.x + threadIdx.x;
    float contrib = 0.f;
    if (idx < NN) {
        int j = idx % NRES, i = idx / NRES;
        float d2 = 1e-12f;
        for (int d = 0; d < 3; ++d) {
            float s1 = 0.f, s2 = 0.f;
            for (int kk = 0; kk < 3; ++kk) {
                s1 += t_r[i * 9 + kk * 3 + d] * (t_t[j * 3 + kk] - t_t[i * 3 + kk]);
                s2 += ttr[i * 9 + kk * 3 + d] * (xt[j * 3 + kk] - ttt[i * 3 + kk]);
            }
            float df = s1 - s2;
            d2 += df * df;
        }
        contrib = fminf(10.f, sqrtf(d2)) * mask[i] * mask[j];
    }
    float bs = block_reduce_sum_256(contrib);
    if (threadIdx.x == 0) atomicAdd(&acc[ACC_F0 + layer], bs);
}

// torsion numerator: sum (ap - at)^2 * m + 0.02 * |l-1| * m  (denominator later)
__global__ void torsion_kernel(const float* __restrict__ alpha,
                               const float* __restrict__ alpha_true,
                               const float* __restrict__ mask,
                               float* __restrict__ acc, int layer)
{
    int idx = blockIdx.x * blockDim.x + threadIdx.x;
    float contrib = 0.f;
    if (idx < 2 * NRES) {
        int n = idx % NRES, g = idx / NRES;
        float a0 = alpha[(g * 2 + 0) * NRES + n];
        float a1 = alpha[(g * 2 + 1) * NRES + n];
        float l  = sqrtf(a0 * a0 + a1 * a1);
        float m  = mask[n];
        float d0 = a0 / l - alpha_true[(g * 2 + 0) * NRES + n];
        float d1 = a1 / l - alpha_true[(g * 2 + 1) * NRES + n];
        contrib = (d0 * d0 + d1 * d1) * m + 0.02f * fabsf(l - 1.f) * m;
    }
    float bs = block_reduce_sum_256(contrib);
    if (threadIdx.x == 0) atomicAdd(&acc[ACC_T0 + layer], bs);
}

// out[0..1151] = final t_t; out[1152] = mean layer loss
__global__ void finalize_kernel(const float* __restrict__ t_t,
                                const float* __restrict__ acc,
                                float* __restrict__ out)
{
    int idx = blockIdx.x * blockDim.x + threadIdx.x;
    if (idx < NRES * 3) out[idx] = t_t[idx];
    if (idx == 0) {
        float msum = acc[ACC_MSUM];
        float denF = fmaxf(msum * msum, 1.f);
        float denT = fmaxf(2.f * msum, 1.f);
        float s = 0.f;
        for (int l = 0; l < NLAYER; ++l)
            s += 0.1f * acc[ACC_F0 + l] / denF + acc[ACC_T0 + l] / denT;
        out[NRES * 3] = s / (float)NLAYER;
    }
}

// ---------------------------------------------------------------------------
// host side
// ---------------------------------------------------------------------------
static inline void launch_gemm(hipStream_t st, const float* A, const float* B,
                               const float* bias, const float* resid, float* C,
                               int M, int N, int K, int reluB, int reluOut)
{
    dim3 grid((N + GEMM_BN - 1) / GEMM_BN, (M + GEMM_BM - 1) / GEMM_BM, 1);
    bool full = (M % GEMM_BM == 0) && (N % GEMM_BN == 0) && (K % GEMM_BK == 0);
    if (full)
        gemm_wmma_kernel<false><<<grid, 128, 0, st>>>(A, B, bias, resid, C,
                                                      M, N, K, reluB, reluOut);
    else
        gemm_wmma_kernel<true><<<grid, 128, 0, st>>>(A, B, bias, resid, C,
                                                     M, N, K, reluB, reluOut);
}

static inline dim3 egrid(long total) { return dim3((unsigned)((total + 255) / 256), 1, 1); }

// workspace layout (in floats)
static constexpr long WS_ZLN  = 0;
static constexpr long WS_BIAS = WS_ZLN  + (long)CZ * NN;
static constexpr long WS_A    = WS_BIAS + (long)NHEAD * NN;
static constexpr long WS_S    = WS_A    + (long)NHEAD * NN;
static constexpr long WS_SIN  = WS_S    + (long)CS * NRES;
static constexpr long WS_Q    = WS_SIN  + (long)CS * NRES;
static constexpr long WS_K    = WS_Q    + (long)CIPA * NHEAD * NRES;
static constexpr long WS_V    = WS_K    + (long)CIPA * NHEAD * NRES;
static constexpr long WS_QP   = WS_V    + (long)CIPA * NHEAD * NRES;
static constexpr long WS_KP   = WS_QP   + (long)3 * NHEAD * NQP * NRES;
static constexpr long WS_VP   = WS_KP   + (long)3 * NHEAD * NQP * NRES;
static constexpr long WS_QG   = WS_VP   + (long)3 * NHEAD * NPV * NRES;
static constexpr long WS_KG   = WS_QG   + (long)3 * NHEAD * NQP * NRES;
static constexpr long WS_VG   = WS_KG   + (long)3 * NHEAD * NQP * NRES;
static constexpr long WS_O3T  = WS_VG   + (long)3 * NHEAD * NPV * NRES;
static constexpr long WS_CAT  = WS_O3T  + (long)3 * NHEAD * NPV * NRES;
static constexpr long WS_H1   = WS_CAT  + (long)INS * NRES;
static constexpr long WS_H2   = WS_H1   + (long)CS * NRES;
static constexpr long WS_AF   = WS_H2   + (long)CS * NRES;
static constexpr long WS_HS   = WS_AF   + (long)CA * NRES;
static constexpr long WS_ALP  = WS_HS   + (long)CA * NRES;
static constexpr long WS_BU   = WS_ALP  + (long)4 * NRES;
static constexpr long WS_TR   = WS_BU   + (long)6 * NRES;
static constexpr long WS_TT   = WS_TR   + (long)NRES * 9;
static constexpr long WS_ACC  = WS_TT   + (long)NRES * 3;

extern "C" void kernel_launch(void* const* d_in, const int* in_sizes, int n_in,
                              void* d_out, int out_size, void* d_ws, size_t ws_size,
                              hipStream_t stream)
{
    (void)in_sizes; (void)n_in; (void)out_size; (void)ws_size;

    // inputs (setup_inputs dict order, params flattened in insertion order)
    const float* in_s_initial = (const float*)d_in[0];
    const float* in_z         = (const float*)d_in[1];
    const float* in_ttrue_r   = (const float*)d_in[2];
    const float* in_ttrue_t   = (const float*)d_in[3];
    const float* in_alpha_t   = (const float*)d_in[4];
    const float* in_x_true    = (const float*)d_in[5];
    const float* in_mask      = (const float*)d_in[6];
    int p = 7;
    const float* ln1_g = (const float*)d_in[p++]; const float* ln1_b = (const float*)d_in[p++];
    const float* lnz_g = (const float*)d_in[p++]; const float* lnz_b = (const float*)d_in[p++];
    const float* ln2_g = (const float*)d_in[p++]; const float* ln2_b = (const float*)d_in[p++];
    const float* ln3_g = (const float*)d_in[p++]; const float* ln3_b = (const float*)d_in[p++];
    const float* Ws1_W = (const float*)d_in[p++]; const float* Ws1_b = (const float*)d_in[p++];
    const float* Wq    = (const float*)d_in[p++];
    const float* Wk    = (const float*)d_in[p++];
    const float* Wv    = (const float*)d_in[p++];
    const float* Wqp   = (const float*)d_in[p++];
    const float* Wkp   = (const float*)d_in[p++];
    const float* Wvp   = (const float*)d_in[p++];
    const float* Wb    = (const float*)d_in[p++];
    const float* gamma = (const float*)d_in[p++];
    const float* Wso   = (const float*)d_in[p++]; const float* bso   = (const float*)d_in[p++];
    const float* T1_W  = (const float*)d_in[p++]; const float* T1_b  = (const float*)d_in[p++];
    const float* T2_W  = (const float*)d_in[p++]; const float* T2_b  = (const float*)d_in[p++];
    const float* T3_W  = (const float*)d_in[p++]; const float* T3_b  = (const float*)d_in[p++];
    const float* A1_W  = (const float*)d_in[p++]; const float* A1_b  = (const float*)d_in[p++];
    const float* A2_W  = (const float*)d_in[p++]; const float* A2_b  = (const float*)d_in[p++];
    const float* S1a_W = (const float*)d_in[p++]; const float* S1a_b = (const float*)d_in[p++];
    const float* S1b_W = (const float*)d_in[p++]; const float* S1b_b = (const float*)d_in[p++];
    const float* S2a_W = (const float*)d_in[p++]; const float* S2a_b = (const float*)d_in[p++];
    const float* S2b_W = (const float*)d_in[p++]; const float* S2b_b = (const float*)d_in[p++];
    const float* A3_W  = (const float*)d_in[p++]; const float* A3_b  = (const float*)d_in[p++];
    const float* BU_W  = (const float*)d_in[p++]; const float* BU_b  = (const float*)d_in[p++];

    float* ws   = (float*)d_ws;
    float* zln  = ws + WS_ZLN;
    float* bias = ws + WS_BIAS;
    float* attn = ws + WS_A;
    float* s    = ws + WS_S;
    float* sin_ = ws + WS_SIN;
    float* q    = ws + WS_Q;
    float* k    = ws + WS_K;
    float* v    = ws + WS_V;
    float* qp   = ws + WS_QP;
    float* kp   = ws + WS_KP;
    float* vp   = ws + WS_VP;
    float* qg   = ws + WS_QG;
    float* kg   = ws + WS_KG;
    float* vg   = ws + WS_VG;
    float* o3t  = ws + WS_O3T;
    float* cat  = ws + WS_CAT;
    float* h1   = ws + WS_H1;
    float* h2   = ws + WS_H2;
    float* af   = ws + WS_AF;
    float* hs   = ws + WS_HS;
    float* alp  = ws + WS_ALP;
    float* bu   = ws + WS_BU;
    float* t_r  = ws + WS_TR;
    float* t_t  = ws + WS_TT;
    float* acc  = ws + WS_ACC;
    float* out  = (float*)d_out;

    // ---- pre-loop (loop-invariant) ----
    zero_acc_kernel<<<1, 32, 0, stream>>>(acc);
    msum_kernel<<<1, 256, 0, stream>>>(in_mask, acc);
    ln_axis_kernel<<<egrid(NRES), 256, 0, stream>>>(in_s_initial, ln1_g, ln1_b, sin_, CS, NRES);
    ln_axis_kernel<<<egrid(NN), 256, 0, stream>>>(in_z, lnz_g, lnz_b, zln, CZ, NN);
    launch_gemm(stream, Ws1_W, sin_, Ws1_b, nullptr, s, CS, NRES, CS, 0, 0);
    init_frames_kernel<<<egrid(NRES), 256, 0, stream>>>(t_r, t_t);
    // pair bias = Wb . z_ln  : M=12, N=147456, K=128
    launch_gemm(stream, Wb, zln, nullptr, nullptr, bias, NHEAD, NN, CZ, 0, 0);

    for (int layer = 0; layer < NLAYER; ++layer) {
        // ---- IPA ----
        launch_gemm(stream, Wq,  s, nullptr, nullptr, q,  CIPA * NHEAD, NRES, CS, 0, 0);
        launch_gemm(stream, Wk,  s, nullptr, nullptr, k,  CIPA * NHEAD, NRES, CS, 0, 0);
        launch_gemm(stream, Wv,  s, nullptr, nullptr, v,  CIPA * NHEAD, NRES, CS, 0, 0);
        launch_gemm(stream, Wqp, s, nullptr, nullptr, qp, 3 * NHEAD * NQP, NRES, CS, 0, 0);
        launch_gemm(stream, Wkp, s, nullptr, nullptr, kp, 3 * NHEAD * NQP, NRES, CS, 0, 0);
        launch_gemm(stream, Wvp, s, nullptr, nullptr, vp, 3 * NHEAD * NPV, NRES, CS, 0, 0);
        apply_frames_kernel<<<egrid(3L * NHEAD * NQP * NRES), 256, 0, stream>>>(qp, t_r, t_t, qg, NQP);
        apply_frames_kernel<<<egrid(3L * NHEAD * NQP * NRES), 256, 0, stream>>>(kp, t_r, t_t, kg, NQP);
        apply_frames_kernel<<<egrid(3L * NHEAD * NPV * NRES), 256, 0, stream>>>(vp, t_r, t_t, vg, NPV);
        logits_kernel<<<egrid((long)NHEAD * NN), 256, 0, stream>>>(q, k, qg, kg, bias, gamma, attn);
        softmax384_kernel<<<NHEAD * NRES, 128, 0, stream>>>(attn);
        o1_kernel<<<egrid((long)CZ * NHEAD * NRES), 256, 0, stream>>>(attn, zln, cat);
        o2_kernel<<<egrid((long)CIPA * NHEAD * NRES), 256, 0, stream>>>(attn, v, cat);
        o3a_kernel<<<egrid(3L * NHEAD * NPV * NRES), 256, 0, stream>>>(attn, vg, o3t);
        o3b_kernel<<<egrid((long)NHEAD * NPV * NRES), 256, 0, stream>>>(o3t, t_r, t_t, cat);
        o3n_kernel<<<egrid(NRES), 256, 0, stream>>>(cat);
        // s = s + Wso . cat + bso
        launch_gemm(stream, Wso, cat, bso, s, s, CS, NRES, INS, 0, 0);
        ln_axis_kernel<<<egrid(NRES), 256, 0, stream>>>(s, ln2_g, ln2_b, s, CS, NRES);
        // ---- transition ----
        launch_gemm(stream, T1_W, s,  T1_b, nullptr, h1, CS, NRES, CS, 0, 1);
        launch_gemm(stream, T2_W, h1, T2_b, nullptr, h2, CS, NRES, CS, 0, 1);
        launch_gemm(stream, T3_W, h2, T3_b, s, s, CS, NRES, CS, 0, 0);
        ln_axis_kernel<<<egrid(NRES), 256, 0, stream>>>(s, ln3_g, ln3_b, s, CS, NRES);
        // ---- backbone update + compose ----
        launch_gemm(stream, BU_W, s, BU_b, nullptr, bu, 6, NRES, CS, 0, 0);
        bb_compose_kernel<<<egrid(NRES), 256, 0, stream>>>(bu, t_r, t_t);
        // ---- angle resnet ----
        launch_gemm(stream, A1_W, s,    A1_b, nullptr, af, CA, NRES, CS, 0, 0);
        launch_gemm(stream, A2_W, sin_, A2_b, af,      af, CA, NRES, CS, 0, 0);
        launch_gemm(stream, S1a_W, af, S1a_b, nullptr, hs, CA, NRES, CA, 1, 1);
        launch_gemm(stream, S1b_W, hs, S1b_b, af,      af, CA, NRES, CA, 0, 0);
        launch_gemm(stream, S2a_W, af, S2a_b, nullptr, hs, CA, NRES, CA, 1, 1);
        launch_gemm(stream, S2b_W, hs, S2b_b, af,      af, CA, NRES, CA, 0, 0);
        launch_gemm(stream, A3_W,  af, A3_b,  nullptr, alp, 4, NRES, CA, 1, 0);
        // ---- losses ----
        fape_kernel<<<egrid(NN), 256, 0, stream>>>(t_r, t_t, in_ttrue_r, in_ttrue_t,
                                                   in_x_true, in_mask, acc, layer);
        torsion_kernel<<<egrid(2 * NRES), 256, 0, stream>>>(alp, in_alpha_t, in_mask, acc, layer);
    }

    finalize_kernel<<<egrid(NRES * 3 + 1), 256, 0, stream>>>(t_t, acc, out);
}
